// SelfAttention_82317343195515
// MI455X (gfx1250) — matrix operand order repaired and checked by reading, not compiled, not run on hardware
//
#include <hip/hip_runtime.h>
#include <math.h>

typedef __attribute__((ext_vector_type(16))) _Float16     v16h;
typedef __attribute__((ext_vector_type(8)))  float        v8f;
typedef __attribute__((ext_vector_type(4)))  unsigned int v4u;
typedef __attribute__((ext_vector_type(4)))  int          v4i;
typedef __attribute__((ext_vector_type(8)))  int          v8i;

#define DIMK   4096
#define NHEAD  32
#define NKVH   8
#define HDIM   128
#define BATCH  32
#define MAXSEQ 2048
#define STARTP 2047
#define TQ     (NHEAD * HDIM)   // 4096
#define TKV    (NKVH * HDIM)    // 1024
#define TCH    64               // timesteps per attention chunk
#define NCHUNK (MAXSEQ / TCH)   // 32

#if defined(__has_builtin)
#if __has_builtin(__builtin_amdgcn_tensor_load_to_lds) && \
    __has_builtin(__builtin_amdgcn_s_wait_tensorcnt)
#define HAVE_TDM 1
#endif
#endif
#ifndef HAVE_TDM
#define HAVE_TDM 0
#endif

// Low 32 bits of a generic pointer to LDS = LDS byte offset (ISA §10.2 aperture).
__device__ __forceinline__ unsigned int lds_off(const void* p)
{
    return (unsigned int)(size_t)p;
}

#if HAVE_TDM
// ---------------------------------------------------------------------------
// TDM 2D tile load: rows x 128 f32 elements, row stride 1024 f32, into LDS.
// Descriptor built per CDNA5 ISA §8 (D# groups 0/1; groups 2/3 zero for 2D).
// ---------------------------------------------------------------------------
__device__ __forceinline__ void tdm_load_2d(const void* gaddr,
                                            unsigned int lds_byte_off,
                                            unsigned int rows)
{
    const unsigned long long ga = (unsigned long long)gaddr;
    v4u g0;
    g0[0] = 1u;                                    // count=1, user mode, no gather
    g0[1] = lds_byte_off;                          // lds_addr (bytes)
    g0[2] = (unsigned int)(ga & 0xffffffffu);      // global_addr[31:0]
    g0[3] = (unsigned int)((ga >> 32) & 0x01ffffffu) | (2u << 30); // ga[56:32], type=2

    v8i g1;
    g1[0] = (2 << 16);                             // workgroup_mask=0, data_size=2 (4B)
    g1[1] = (int)(128u << 16);                     // tensor_dim0[15:0] = 128
    g1[2] = (int)(2048u << 16);                    // tensor_dim0 hi=0 | tensor_dim1 lo = 2048
    g1[3] = (int)(128u << 16);                     // tensor_dim1 hi=0 | tile_dim0 = 128
    g1[4] = (int)(rows & 0xffffu);                 // tile_dim1 = rows, tile_dim2 = 0
    g1[5] = (int)1024;                             // tensor_dim0_stride[31:0] (elements)
    g1[6] = 0;                                     // stride hi / dim1_stride lo
    g1[7] = 0;

    const v4i z4 = {0, 0, 0, 0};
#if __clang_major__ >= 23
    const v8i z8 = {0, 0, 0, 0, 0, 0, 0, 0};
    __builtin_amdgcn_tensor_load_to_lds(g0, g1, z4, z4, z8, 0);
#else
    __builtin_amdgcn_tensor_load_to_lds(g0, g1, z4, z4, 0);
#endif
}
#endif

// ---------------------------------------------------------------------------
// Block GEMM: C[32 x 64] += X[32 x 4096] * W[4096 x ldw] (64-col slab at
// wcol0). Weight tiles (32 k-rows x 64 cols f32) staged via coalesced b128
// loads into LDS; per-wave WMMA B fragments gathered from LDS.
// Fragment layouts per ISA §7.12.2 (16-bit A 16x32, 16-bit B 32x16, f32 C).
// ---------------------------------------------------------------------------
__device__ __forceinline__ void wmma_gemm_block(
    const float* __restrict__ X,
    const float* __restrict__ W, int ldw, int wcol0,
    float* wtile, v8f& c0, v8f& c1)
{
    const int tid  = threadIdx.x;        // 0..127
    const int wave = tid >> 5;
    const int lane = tid & 31;
    const int hl   = lane >> 4;
    const int mrow = lane & 15;

#pragma unroll
    for (int i = 0; i < 8; ++i) { c0[i] = 0.0f; c1[i] = 0.0f; }

    for (int kk = 0; kk < DIMK; kk += 32) {
        // cooperative stage: 32x64 f32 tile, fully coalesced 128-bit loads
#pragma unroll
        for (int rep = 0; rep < 4; ++rep) {
            const int idx = rep * 512 + tid * 4;
            const int r   = idx >> 6;
            const int cl  = idx & 63;
            *reinterpret_cast<float4*>(&wtile[idx]) =
                *reinterpret_cast<const float4*>(
                    &W[(size_t)(kk + r) * ldw + wcol0 + cl]);
        }
        if (kk + 32 < DIMK)
            __builtin_prefetch(&W[(size_t)(kk + 32 + (tid >> 2)) * ldw +
                                  wcol0 + ((tid & 3) << 4)], 0, 1);
        __syncthreads();

        v16h a0, a1, b;
#pragma unroll
        for (int i = 0; i < 16; ++i) {
            const int k = kk + 8 * hl + ((i < 8) ? i : (i + 8));
            a0[i] = (_Float16)X[(size_t)mrow * DIMK + k];
            a1[i] = (_Float16)X[(size_t)(mrow + 16) * DIMK + k];
        }
#pragma unroll
        for (int i = 0; i < 16; ++i)
            b[i] = (_Float16)wtile[(8 * hl + ((i < 8) ? i : (i + 8))) * 64 + wave * 16 + mrow];

        c0 = __builtin_amdgcn_wmma_f32_16x16x32_f16(false, a0, false, b,
                                                    (short)0, c0, false, false);
        c1 = __builtin_amdgcn_wmma_f32_16x16x32_f16(false, a1, false, b,
                                                    (short)0, c1, false, false);
        __syncthreads();
    }
}

// RoPE on a C fragment: even/odd column pair lives in lane and lane^1.
__device__ __forceinline__ void rope_tile(v8f& c, int lane, float cs, float sn)
{
    const bool odd = (lane & 1) != 0;
#pragma unroll
    for (int j = 0; j < 8; ++j) {
        const float v = c[j];
        const float p = __shfl_xor(v, 1, 32);
        c[j] = odd ? (p * sn + v * cs) : (v * cs - p * sn);
    }
}

// ---------------------------------------------------------------------------
// Kernel 1: fused QKV projection + RoPE(q,k). Block = 64-col slab of the
// combined [wq | wk | wv] weight block (boundaries are multiples of 64).
// ---------------------------------------------------------------------------
__global__ void qkv_rope_kernel(const float* __restrict__ x,
                                const float* __restrict__ wq,
                                const float* __restrict__ wk,
                                const float* __restrict__ wv,
                                const float* __restrict__ fc,
                                const float* __restrict__ fs,
                                float* __restrict__ q_ws,
                                float* __restrict__ k_ws,
                                float* __restrict__ v_ws)
{
    __shared__ float wtile[32 * 64];

    const int wave = threadIdx.x >> 5;
    const int lane = threadIdx.x & 31;
    const int hl   = lane >> 4;
    const int cb0  = blockIdx.x * 64;

    const float* W; float* dst; int ldw, ldd, wcol0; bool rope;
    if (cb0 < TQ)            { W = wq; ldw = TQ;  wcol0 = cb0;            dst = q_ws; ldd = TQ;  rope = true;  }
    else if (cb0 < TQ + TKV) { W = wk; ldw = TKV; wcol0 = cb0 - TQ;       dst = k_ws; ldd = TKV; rope = true;  }
    else                     { W = wv; ldw = TKV; wcol0 = cb0 - TQ - TKV; dst = v_ws; ldd = TKV; rope = false; }

    v8f c0, c1;
    wmma_gemm_block(x, W, ldw, wcol0, wtile, c0, c1);

    const int ncol = wcol0 + wave * 16 + (lane & 15);
    if (rope) {
        const int d = ncol & (HDIM - 1);
        const float cs = fc[d >> 1];
        const float sn = fs[d >> 1];
        rope_tile(c0, lane, cs, sn);
        rope_tile(c1, lane, cs, sn);
    }
#pragma unroll
    for (int j = 0; j < 8; ++j) {
        dst[(size_t)(j + 8 * hl) * ldd + ncol]      = c0[j];
        dst[(size_t)(16 + j + 8 * hl) * ldd + ncol] = c1[j];
    }
}

// ---------------------------------------------------------------------------
// Kernel 2: flash-decoding attention. Block = (batch b, kv-head g).
// K/V streamed into LDS in 64x128 f32 tiles via the Tensor Data Mover
// (double-buffered, TENSORcnt-synchronized); 8 waves each cover 8 rows per
// tile; 4 Q-heads share the K/V stream. t == 2047 comes from workspace.
// ---------------------------------------------------------------------------
__global__ void attn_kernel(const float* __restrict__ q_ws,
                            const float* __restrict__ k_ws,
                            const float* __restrict__ v_ws,
                            const float* __restrict__ k_cache,
                            const float* __restrict__ v_cache,
                            float* __restrict__ a_ws)
{
    extern __shared__ float smem[];
    float* lk   = smem;                      // 2 * TCH * HDIM
    float* lv   = lk + 2 * TCH * HDIM;       // 2 * TCH * HDIM
    float* sm_  = lv + 2 * TCH * HDIM;       // 8*4
    float* sl_  = sm_ + 32;                  // 8*4
    float* sacc = sl_ + 32;                  // 8*4*HDIM

    const int b    = blockIdx.x;
    const int g    = blockIdx.y;
    const int tid  = threadIdx.x;
    const int lane = tid & 31;
    const int w    = tid >> 5;               // 0..7
    const float scale = 0.088388347648318447f;   // 1/sqrt(128)

    const float* kbase = k_cache + (size_t)b * MAXSEQ * TKV + g * HDIM;
    const float* vbase = v_cache + (size_t)b * MAXSEQ * TKV + g * HDIM;

    float4 q[4];
#pragma unroll
    for (int r = 0; r < 4; ++r)
        q[r] = *reinterpret_cast<const float4*>(
            q_ws + (size_t)b * TQ + (g * 4 + r) * HDIM + lane * 4);

    float  m[4], l[4];
    float4 acc[4];
#pragma unroll
    for (int r = 0; r < 4; ++r) {
        m[r] = -INFINITY; l[r] = 0.0f;
        acc[r] = make_float4(0.0f, 0.0f, 0.0f, 0.0f);
    }

#if HAVE_TDM
    if (w == 0) {
        tdm_load_2d(kbase, lds_off(lk), TCH);
        tdm_load_2d(vbase, lds_off(lv), TCH);
    }
#endif

    for (int c = 0; c < NCHUNK; ++c) {
        const int cur = c & 1;
        float* lkc = lk + cur * TCH * HDIM;
        float* lvc = lv + cur * TCH * HDIM;

#if HAVE_TDM
        if (w == 0) {
            if (c + 1 < NCHUNK) {
                tdm_load_2d(kbase + (size_t)(c + 1) * TCH * TKV,
                            lds_off(lk + (cur ^ 1) * TCH * HDIM), TCH);
                tdm_load_2d(vbase + (size_t)(c + 1) * TCH * TKV,
                            lds_off(lv + (cur ^ 1) * TCH * HDIM), TCH);
                __builtin_amdgcn_s_wait_tensorcnt(2);   // current tile done
            } else {
                __builtin_amdgcn_s_wait_tensorcnt(0);
            }
        }
        __syncthreads();
#else
        {
            const float* kc0 = kbase + (size_t)c * TCH * TKV;
            const float* vc0 = vbase + (size_t)c * TCH * TKV;
#pragma unroll
            for (int rep = 0; rep < 8; ++rep) {
                const int idx = rep * 1024 + tid * 4;
                const int r   = idx >> 7;
                const int cl  = idx & 127;
                *reinterpret_cast<float4*>(&lkc[idx]) =
                    *reinterpret_cast<const float4*>(&kc0[(size_t)r * TKV + cl]);
                *reinterpret_cast<float4*>(&lvc[idx]) =
                    *reinterpret_cast<const float4*>(&vc0[(size_t)r * TKV + cl]);
            }
        }
        __syncthreads();
#endif

        if (c == NCHUNK - 1) {          // patch t=2047 with fresh rope'd k/v
            if (tid < HDIM)
                lkc[(TCH - 1) * HDIM + tid] =
                    k_ws[(size_t)b * TKV + g * HDIM + tid];
            else if (tid < 2 * HDIM)
                lvc[(TCH - 1) * HDIM + (tid - HDIM)] =
                    v_ws[(size_t)b * TKV + g * HDIM + (tid - HDIM)];
            __syncthreads();
        }

#pragma unroll
        for (int j = 0; j < 8; ++j) {
            const int tl = w * 8 + j;
            const float4 kv = *reinterpret_cast<const float4*>(&lkc[tl * HDIM + lane * 4]);
            const float4 vv = *reinterpret_cast<const float4*>(&lvc[tl * HDIM + lane * 4]);

            float s[4];
#pragma unroll
            for (int r = 0; r < 4; ++r)
                s[r] = q[r].x * kv.x + q[r].y * kv.y + q[r].z * kv.z + q[r].w * kv.w;
#pragma unroll
            for (int off = 16; off > 0; off >>= 1) {
#pragma unroll
                for (int r = 0; r < 4; ++r)
                    s[r] += __shfl_xor(s[r], off, 32);
            }
#pragma unroll
            for (int r = 0; r < 4; ++r) {
                const float sc    = s[r] * scale;
                const float mn    = fmaxf(m[r], sc);
                const float alpha = __expf(m[r] - mn);
                const float p     = __expf(sc - mn);
                l[r] = l[r] * alpha + p;
                acc[r].x = acc[r].x * alpha + p * vv.x;
                acc[r].y = acc[r].y * alpha + p * vv.y;
                acc[r].z = acc[r].z * alpha + p * vv.z;
                acc[r].w = acc[r].w * alpha + p * vv.w;
                m[r] = mn;
            }
        }
        __syncthreads();                // buffer may be overwritten next iter
    }

    // combine 8 wave partials
    if (lane == 0) {
#pragma unroll
        for (int r = 0; r < 4; ++r) { sm_[w * 4 + r] = m[r]; sl_[w * 4 + r] = l[r]; }
    }
#pragma unroll
    for (int r = 0; r < 4; ++r)
        *reinterpret_cast<float4*>(&sacc[(w * 4 + r) * HDIM + lane * 4]) = acc[r];
    __syncthreads();

#pragma unroll
    for (int i = 0; i < 2; ++i) {
        const int idx = tid + i * 256;
        const int r   = idx >> 7;
        const int d   = idx & (HDIM - 1);
        float M = -INFINITY;
#pragma unroll
        for (int ww = 0; ww < 8; ++ww) M = fmaxf(M, sm_[ww * 4 + r]);
        float L = 0.0f, V = 0.0f;
#pragma unroll
        for (int ww = 0; ww < 8; ++ww) {
            const float e = __expf(sm_[ww * 4 + r] - M);
            L += sl_[ww * 4 + r] * e;
            V += sacc[(ww * 4 + r) * HDIM + d] * e;
        }
        a_ws[(size_t)b * TQ + (g * 4 + r) * HDIM + d] = V / L;
    }
}

// ---------------------------------------------------------------------------
// Kernel 3: output projection, attn[32 x 4096] * wo[4096 x 4096] -> out.
// ---------------------------------------------------------------------------
__global__ void oproj_kernel(const float* __restrict__ a_ws,
                             const float* __restrict__ wo,
                             float* __restrict__ out)
{
    __shared__ float wtile[32 * 64];

    const int wave = threadIdx.x >> 5;
    const int lane = threadIdx.x & 31;
    const int hl   = lane >> 4;
    const int cb0  = blockIdx.x * 64;

    v8f c0, c1;
    wmma_gemm_block(a_ws, wo, DIMK, cb0, wtile, c0, c1);

    const int ncol = cb0 + wave * 16 + (lane & 15);
#pragma unroll
    for (int j = 0; j < 8; ++j) {
        out[(size_t)(j + 8 * hl) * DIMK + ncol]      = c0[j];
        out[(size_t)(16 + j + 8 * hl) * DIMK + ncol] = c1[j];
    }
}

// ---------------------------------------------------------------------------
extern "C" void kernel_launch(void* const* d_in, const int* in_sizes, int n_in,
                              void* d_out, int out_size, void* d_ws, size_t ws_size,
                              hipStream_t stream)
{
    const float* x  = (const float*)d_in[0];
    const float* wq = (const float*)d_in[1];
    const float* wk = (const float*)d_in[2];
    const float* wv = (const float*)d_in[3];
    const float* wo = (const float*)d_in[4];
    const float* kc = (const float*)d_in[5];
    const float* vc = (const float*)d_in[6];
    const float* fc = (const float*)d_in[7];
    const float* fs = (const float*)d_in[8];
    float* out = (float*)d_out;

    float* q_ws = (float*)d_ws;              // 32*4096
    float* k_ws = q_ws + BATCH * TQ;         // 32*1024
    float* v_ws = k_ws + BATCH * TKV;        // 32*1024
    float* a_ws = v_ws + BATCH * TKV;        // 32*4096

    // 6144 cols / 64 per block = 96 blocks
    qkv_rope_kernel<<<dim3(96), dim3(128), 0, stream>>>(
        x, wq, wk, wv, fc, fs, q_ws, k_ws, v_ws);

    // dynamic LDS: K/V double buffers + softmax combine scratch
    const size_t smem_bytes =
        (size_t)(2 * TCH * HDIM * 2 + 32 + 32 + 8 * 4 * HDIM) * sizeof(float);
    attn_kernel<<<dim3(BATCH, NKVH), dim3(256), smem_bytes, stream>>>(
        q_ws, k_ws, v_ws, kc, vc, a_ws);

    // 4096 cols / 64 per block = 64 blocks
    oproj_kernel<<<dim3(64), dim3(128), 0, stream>>>(a_ws, wo, out);
}
